// GeoSSL_PDM_68796786147703
// MI455X (gfx1250) — matrix operand order repaired and checked by reading, not compiled, not run on hardware
//
#include <hip/hip_runtime.h>
#include <hip/hip_bf16.h>

// ---------------------------------------------------------------------------
// GeoSSL PDM loss for MI455X (gfx1250, wave32, WMMA).
//   Part A: per-graph Riemann score target + per-node MSE, segment-reduced.
//           One wave32 per graph (node2graph is sorted -> binary search bounds).
//   Part B: CE head  logits = silu(M@W1+b1)@W2+b2 ; fused 32-row-tile WMMA
//           pipeline (f16 inputs, f32 accum). Each wave computes TWO 16-row
//           M-tiles per loaded B panel -> halves W traffic from L2 and doubles
//           WMMA work per loaded byte vs a 16-row tile.
// ---------------------------------------------------------------------------

typedef __attribute__((ext_vector_type(16))) _Float16 v16h;
typedef __attribute__((ext_vector_type(8)))  _Float16 v8h;
typedef __attribute__((ext_vector_type(8)))  float    v8f;

#define D_DIM   300
#define KPAD    320     // K padded to 10 x 32
#define NPAD    304     // N padded to 19 x 16
#define KSTEPS  10
#define NTILES  19
#define ROWS    32      // rows (graphs) per block in Part B

// ------------------------------ helpers ------------------------------------

static __device__ __forceinline__ float wave_red(float v) {
#pragma unroll
  for (int m = 16; m > 0; m >>= 1) v += __shfl_xor(v, m, 32);
  return v;
}

static __device__ __forceinline__ int lower_bound_i(const int* __restrict__ a,
                                                    int n, int key) {
  int lo = 0, hi = n;
  while (lo < hi) {
    int mid = (lo + hi) >> 1;
    if (a[mid] < key) lo = mid + 1; else hi = mid;
  }
  return lo;
}

static __device__ __forceinline__ v16h cat16(v8h lo, v8h hi) {
  v16h r;
#pragma unroll
  for (int i = 0; i < 8; ++i) { r[i] = lo[i]; r[i + 8] = hi[i]; }
  return r;
}

// ------------------------- Part A: one wave per graph ----------------------

__global__ __launch_bounds__(256)
void partA_kernel(const int* __restrict__ n2g,
                  const int* __restrict__ noise_level,
                  const float* __restrict__ pos_noise_pred,
                  const float* __restrict__ pos_perturbed,
                  const float* __restrict__ pos_target,
                  const float* __restrict__ sigmas,
                  float* __restrict__ partialA,
                  int N, int G)
{
  __shared__ float wsum[8];
  const int wave = threadIdx.x >> 5;
  const int lane = threadIdx.x & 31;
  const int g = blockIdx.x * 8 + wave;

  float per_graph = 0.f;
  if (g < G) {
    const int start = lower_bound_i(n2g, N, g);
    const int end   = lower_bound_i(n2g, N, g + 1);
    const int cnt   = end - start;
    if (cnt > 0) {
      // pass 1: segment means of pos_target / pos_perturbed
      float st0 = 0.f, st1 = 0.f, st2 = 0.f;
      float sp0 = 0.f, sp1 = 0.f, sp2 = 0.f;
      for (int i = start + lane; i < end; i += 32) {
        st0 += pos_target[3 * i + 0];
        st1 += pos_target[3 * i + 1];
        st2 += pos_target[3 * i + 2];
        sp0 += pos_perturbed[3 * i + 0];
        sp1 += pos_perturbed[3 * i + 1];
        sp2 += pos_perturbed[3 * i + 2];
      }
      st0 = wave_red(st0); st1 = wave_red(st1); st2 = wave_red(st2);
      sp0 = wave_red(sp0); sp1 = wave_red(sp1); sp2 = wave_red(sp2);
      const float inv_cnt = 1.f / (float)cnt;
      const float c0 = st0 * inv_cnt, c1 = st1 * inv_cnt, c2 = st2 * inv_cnt;
      const float q0 = sp0 * inv_cnt, q1 = sp1 * inv_cnt, q2 = sp2 * inv_cnt;

      // pass 2: per-graph 3x3 matrices ptp, otp
      float ptp[9], otp[9];
#pragma unroll
      for (int j = 0; j < 9; ++j) { ptp[j] = 0.f; otp[j] = 0.f; }
      for (int i = start + lane; i < end; i += 32) {
        const float pp0 = pos_perturbed[3 * i + 0] - q0;
        const float pp1 = pos_perturbed[3 * i + 1] - q1;
        const float pp2 = pos_perturbed[3 * i + 2] - q2;
        const float pt0 = pos_target[3 * i + 0] - c0;
        const float pt1 = pos_target[3 * i + 1] - c1;
        const float pt2 = pos_target[3 * i + 2] - c2;
        const float pp[3] = { pp0, pp1, pp2 };
        const float pt[3] = { pt0, pt1, pt2 };
#pragma unroll
        for (int r = 0; r < 3; ++r)
#pragma unroll
          for (int cidx = 0; cidx < 3; ++cidx) {
            ptp[r * 3 + cidx] += pp[r] * pp[cidx];
            otp[r * 3 + cidx] += pt[r] * pp[cidx];
          }
      }
#pragma unroll
      for (int j = 0; j < 9; ++j) {
        ptp[j] = wave_red(ptp[j]);
        otp[j] = wave_red(otp[j]);
      }
      float fp = 0.f, fo = 0.f;
#pragma unroll
      for (int j = 0; j < 9; ++j) { fp += ptp[j] * ptp[j]; fo += otp[j] * otp[j]; }
      const float den = sqrtf(fp) + sqrtf(fo);

      const float sig = sigmas[noise_level[g]];
      const float inv_sig = 1.f / sig;
      const float inv_den_sig = inv_sig / den;

      // pass 3: per-node loss
      float acc = 0.f;
      for (int i = start + lane; i < end; i += 32) {
        const float pp0 = pos_perturbed[3 * i + 0] - q0;
        const float pp1 = pos_perturbed[3 * i + 1] - q1;
        const float pp2 = pos_perturbed[3 * i + 2] - q2;
        const float pt0 = pos_target[3 * i + 0] - c0;
        const float pt1 = pos_target[3 * i + 1] - c1;
        const float pt2 = pos_target[3 * i + 2] - c2;
#pragma unroll
        for (int j = 0; j < 3; ++j) {
          const float num = -2.f * ((pp0 * ptp[0 + j] + pp1 * ptp[3 + j] + pp2 * ptp[6 + j]) -
                                    (pt0 * otp[0 + j] + pt1 * otp[3 + j] + pt2 * otp[6 + j]));
          const float score = num * inv_den_sig;
          const float pred = (pos_noise_pred[3 * i + j] - pos_perturbed[3 * i + j]) * inv_sig;
          const float d = pred - score;
          acc += d * d;
        }
      }
      per_graph = wave_red(acc);
    }
  }
  if (lane == 0) wsum[wave] = per_graph;
  __syncthreads();
  if (threadIdx.x == 0) {
    float s = 0.f;
#pragma unroll
    for (int i = 0; i < 8; ++i) s += wsum[i];
    partialA[blockIdx.x] = s;
  }
}

// ------------------- Part B prep: f16 transposed/padded weights ------------

__global__ __launch_bounds__(256)
void prep_weights(const float* __restrict__ W1, const float* __restrict__ b1,
                  const float* __restrict__ W2, const float* __restrict__ b2,
                  _Float16* __restrict__ W1t, _Float16* __restrict__ W2t,
                  float* __restrict__ b1p, float* __restrict__ b2p)
{
  const int idx = blockIdx.x * blockDim.x + threadIdx.x;
  if (idx < NPAD * KPAD) {
    const int n = idx / KPAD;
    const int k = idx % KPAD;
    const bool ok = (n < D_DIM) && (k < D_DIM);
    W1t[idx] = ok ? (_Float16)W1[k * D_DIM + n] : (_Float16)0.f;
    W2t[idx] = ok ? (_Float16)W2[k * D_DIM + n] : (_Float16)0.f;
  }
  if (idx < NPAD) {
    b1p[idx] = (idx < D_DIM) ? b1[idx] : 0.f;
    b2p[idx] = (idx < D_DIM) ? b2[idx] : 0.f;
  }
}

// ---------------- Part B: fused MLP + CE, 32 rows per block ----------------
// Block = 128 threads (4 wave32). Two 16-row M-tiles per block; each wave
// reuses every loaded B panel (one N-tile, one K-step) for two WMMAs.

__global__ __launch_bounds__(128)
void mlp_ce_kernel(const float* __restrict__ M,
                   const int* __restrict__ noise_level,
                   const _Float16* __restrict__ W1t,
                   const _Float16* __restrict__ W2t,
                   const float* __restrict__ b1p,
                   const float* __restrict__ b2p,
                   float* __restrict__ partialB,
                   int G)
{
  __shared__ __align__(16) _Float16 Atile[ROWS * KPAD];   // 20 KB
  __shared__ __align__(16) _Float16 Htile[ROWS * KPAD];   // 20 KB
  __shared__ __align__(16) float    Ltile[ROWS * NPAD];   // 38.9 KB
  __shared__ float ceR[ROWS];

  const int tid = threadIdx.x;
  const int m0 = blockIdx.x * ROWS;

  // Load + convert A tile (fp32 -> f16), zero K padding.
  for (int idx = tid; idx < ROWS * KPAD; idx += 128) {
    const int r = idx / KPAD;
    const int c = idx % KPAD;
    float v = 0.f;
    if (c < D_DIM && (m0 + r) < G) v = M[(size_t)(m0 + r) * D_DIM + c];
    Atile[idx] = (_Float16)v;
  }
  // Zero H K-padding (cols 304..319) once; never written by GEMM1.
  for (int idx = tid; idx < ROWS * (KPAD - NPAD); idx += 128) {
    const int r = idx / (KPAD - NPAD);
    const int c = NPAD + idx % (KPAD - NPAD);
    Htile[r * KPAD + c] = (_Float16)0.f;
  }
  __syncthreads();

  const int wave  = tid >> 5;
  const int lane  = tid & 31;
  const int rowA  = lane & 15;        // A row within a 16-row M-tile
  const int khalf = lane >> 4;        // 0: K 0..7/16..23, 1: K 8..15/24..31

  // ---- GEMM1: H = silu(A @ W1 + b1) ----
  for (int nt = wave; nt < NTILES; nt += 4) {
    v8f acc0 = {};
    v8f acc1 = {};
    const int ncol = nt * 16 + (lane & 15);
    const _Float16* bbase = W1t + (size_t)ncol * KPAD;
#pragma unroll
    for (int kt = 0; kt < KSTEPS; ++kt) {
      const int kb = kt * 32;
      v8h blo = *(const v8h*)&bbase[kb + khalf * 8];
      v8h bhi = *(const v8h*)&bbase[kb + khalf * 8 + 16];
      const v16h b = cat16(blo, bhi);
      // M-tile 0 (rows 0..15)
      v8h a0lo = *(const v8h*)&Atile[rowA * KPAD + kb + khalf * 8];
      v8h a0hi = *(const v8h*)&Atile[rowA * KPAD + kb + khalf * 8 + 16];
      acc0 = __builtin_amdgcn_wmma_f32_16x16x32_f16(
          false, cat16(a0lo, a0hi), false, b, (short)0, acc0, false, false);
      // M-tile 1 (rows 16..31)
      v8h a1lo = *(const v8h*)&Atile[(16 + rowA) * KPAD + kb + khalf * 8];
      v8h a1hi = *(const v8h*)&Atile[(16 + rowA) * KPAD + kb + khalf * 8 + 16];
      acc1 = __builtin_amdgcn_wmma_f32_16x16x32_f16(
          false, cat16(a1lo, a1hi), false, b, (short)0, acc1, false, false);
    }
    const float bias = b1p[ncol];
#pragma unroll
    for (int r = 0; r < 8; ++r) {
      const int mrow = r + khalf * 8;
      float x0 = acc0[r] + bias;
      float x1 = acc1[r] + bias;
      Htile[mrow * KPAD + ncol]        = (_Float16)(x0 / (1.f + __expf(-x0)));
      Htile[(16 + mrow) * KPAD + ncol] = (_Float16)(x1 / (1.f + __expf(-x1)));
    }
  }
  __syncthreads();

  // ---- GEMM2: logits = H @ W2 + b2 ----
  for (int nt = wave; nt < NTILES; nt += 4) {
    v8f acc0 = {};
    v8f acc1 = {};
    const int ncol = nt * 16 + (lane & 15);
    const _Float16* bbase = W2t + (size_t)ncol * KPAD;
#pragma unroll
    for (int kt = 0; kt < KSTEPS; ++kt) {
      const int kb = kt * 32;
      v8h blo = *(const v8h*)&bbase[kb + khalf * 8];
      v8h bhi = *(const v8h*)&bbase[kb + khalf * 8 + 16];
      const v16h b = cat16(blo, bhi);
      v8h a0lo = *(const v8h*)&Htile[rowA * KPAD + kb + khalf * 8];
      v8h a0hi = *(const v8h*)&Htile[rowA * KPAD + kb + khalf * 8 + 16];
      acc0 = __builtin_amdgcn_wmma_f32_16x16x32_f16(
          false, cat16(a0lo, a0hi), false, b, (short)0, acc0, false, false);
      v8h a1lo = *(const v8h*)&Htile[(16 + rowA) * KPAD + kb + khalf * 8];
      v8h a1hi = *(const v8h*)&Htile[(16 + rowA) * KPAD + kb + khalf * 8 + 16];
      acc1 = __builtin_amdgcn_wmma_f32_16x16x32_f16(
          false, cat16(a1lo, a1hi), false, b, (short)0, acc1, false, false);
    }
    const float bias = b2p[ncol];
#pragma unroll
    for (int r = 0; r < 8; ++r) {
      const int mrow = r + khalf * 8;
      Ltile[mrow * NPAD + ncol]        = acc0[r] + bias;
      Ltile[(16 + mrow) * NPAD + ncol] = acc1[r] + bias;
    }
  }
  __syncthreads();

  // ---- Cross-entropy per row (pad cols 300..303 excluded) ----
  if (tid < ROWS) {
    const int row = tid;
    float ce = 0.f;
    if ((m0 + row) < G) {
      const float* lr = &Ltile[row * NPAD];
      float mx = -3.0e38f;
      for (int c = 0; c < D_DIM; ++c) mx = fmaxf(mx, lr[c]);
      float se = 0.f;
      for (int c = 0; c < D_DIM; ++c) se += __expf(lr[c] - mx);
      const int tgt = noise_level[m0 + row];
      ce = (mx + __logf(se)) - lr[tgt];
    }
    ceR[row] = ce;
  }
  __syncthreads();
  if (tid == 0) {
    float s = 0.f;
#pragma unroll
    for (int i = 0; i < ROWS; ++i) s += ceR[i];
    partialB[blockIdx.x] = s;
  }
}

// ------------------------------ final reduce -------------------------------

__global__ __launch_bounds__(256)
void final_reduce(const float* __restrict__ partialA, int nA,
                  const float* __restrict__ partialB, int nB,
                  float invG, float* __restrict__ out)
{
  __shared__ float sh[256];
  const int tid = threadIdx.x;
  float sa = 0.f, sb = 0.f;
  for (int i = tid; i < nA; i += 256) sa += partialA[i];
  for (int i = tid; i < nB; i += 256) sb += partialB[i];

  sh[tid] = sa; __syncthreads();
  for (int s = 128; s > 0; s >>= 1) {
    if (tid < s) sh[tid] += sh[tid + s];
    __syncthreads();
  }
  if (tid == 0) out[0] = sh[0] * invG;
  __syncthreads();

  sh[tid] = sb; __syncthreads();
  for (int s = 128; s > 0; s >>= 1) {
    if (tid < s) sh[tid] += sh[tid + s];
    __syncthreads();
  }
  if (tid == 0) out[1] = sh[0] * invG;
}

// ------------------------------- launcher ----------------------------------

extern "C" void kernel_launch(void* const* d_in, const int* in_sizes, int n_in,
                              void* d_out, int out_size, void* d_ws, size_t ws_size,
                              hipStream_t stream)
{
  const int*   node2graph     = (const int*)d_in[0];
  /* d_in[1] edge_index: unused by reference output */
  const int*   noise_level    = (const int*)d_in[2];
  /* d_in[3] num_graphs scalar: use in_sizes instead */
  const float* molecule_repr  = (const float*)d_in[4];
  const float* pos_noise_pred = (const float*)d_in[5];
  const float* pos_perturbed  = (const float*)d_in[6];
  const float* pos_target     = (const float*)d_in[7];
  const float* W1             = (const float*)d_in[8];
  const float* b1             = (const float*)d_in[9];
  const float* W2             = (const float*)d_in[10];
  const float* b2             = (const float*)d_in[11];
  const float* sigmas         = (const float*)d_in[12];

  const int N = in_sizes[0];   // 2,000,000
  const int G = in_sizes[2];   // 100,000

  // Workspace layout (all fully overwritten each call; ~450 KB total).
  _Float16* W1t = (_Float16*)d_ws;
  _Float16* W2t = W1t + (size_t)NPAD * KPAD;
  float* b1p = (float*)(W2t + (size_t)NPAD * KPAD);
  float* b2p = b1p + NPAD;
  float* partialA = b2p + NPAD;
  const int nblkA = (G + 7) / 8;              // 12,500 blocks, 8 graphs each
  float* partialB = partialA + nblkA;
  const int nblkB = (G + ROWS - 1) / ROWS;    // 3,125 row-tiles

  prep_weights<<<(NPAD * KPAD + 255) / 256, 256, 0, stream>>>(
      W1, b1, W2, b2, W1t, W2t, b1p, b2p);

  partA_kernel<<<nblkA, 256, 0, stream>>>(
      node2graph, noise_level, pos_noise_pred, pos_perturbed, pos_target,
      sigmas, partialA, N, G);

  mlp_ce_kernel<<<nblkB, 128, 0, stream>>>(
      molecule_repr, noise_level, W1t, W2t, b1p, b2p, partialB, G);

  final_reduce<<<1, 256, 0, stream>>>(
      partialA, nblkA, partialB, nblkB, 1.0f / (float)G, (float*)d_out);
}
